// NPTransitionPrior_43465069035864
// MI455X (gfx1250) — matrix operand order repaired and checked by reading, not compiled, not run on hardware
//
#include <hip/hip_runtime.h>
#include <hip/hip_bf16.h>
#include <math.h>

#define SLOPE 0.2f

typedef __attribute__((ext_vector_type(16))) __bf16        v16bf;
typedef __attribute__((ext_vector_type(8)))  __bf16        bf16x8;
typedef __attribute__((ext_vector_type(8)))  float         v8f;
typedef __attribute__((ext_vector_type(8)))  unsigned int  u32x8;

// ---------------------------------------------------------------------------
// D = A(16x32 bf16) * B(32x16 bf16) + C(16x16 f32)   -- wave32 CDNA5 WMMA
// B columns 0-7 carry 8 samples' primal, columns 8-15 the same samples' JVP
// tangent, so ONE accumulator set serves forward + derivative.
// ---------------------------------------------------------------------------
__device__ __forceinline__ v8f wmma_bf16(v16bf a, v16bf b, v8f c) {
  return __builtin_amdgcn_wmma_f32_16x16x32_bf16(
      /*neg_a=*/false, a, /*neg_b=*/false, b,
      /*c_mod=*/(short)0, c, /*reuse_a=*/false, /*reuse_b=*/false);
}

// Load one 16x32 bf16 A-tile from row-major LDS [128 x stride].
// A layout: lane(0-15) row=lane, v0..7 = K(16h .. 16h+15) (contiguous)
__device__ __forceinline__ v16bf load_A(const __bf16* Wl, int stride, int m,
                                        int kbase, int row, int half) {
  const __bf16* p = Wl + (m * 16 + row) * stride + kbase + half * 16;
  bf16x8 lo = *(const bf16x8*)p;           // 16 B
  bf16x8 hi = *(const bf16x8*)(p + 8);     // +16 B
  return __builtin_shufflevector(lo, hi, 0, 1, 2, 3, 4, 5, 6, 7,
                                 8, 9, 10, 11, 12, 13, 14, 15);
}

__device__ __forceinline__ unsigned int pack_bf16(float a, float b) {
  union { __bf16 h[2]; unsigned int u; } cv;
  cv.h[0] = (__bf16)a;   // low 16 bits = even element
  cv.h[1] = (__bf16)b;
  return cv.u;
}

// LeakyReLU: primal cols get bias+lrelu from own sign; tangent cols get the
// derivative mask from the partner primal lane (lane ^ 8), exchanged as one
// packed 8-bit sign mask per m-tile.
__device__ __forceinline__ void act_bias_mixed(v8f C[8], const float* bias,
                                               int half, bool isH) {
#pragma unroll
  for (int m = 0; m < 8; ++m) {
    const float* bp = bias + m * 16 + half * 8;
    float4 blo = *(const float4*)bp;
    float4 bhi = *(const float4*)(bp + 4);
    float bb[8] = {blo.x, blo.y, blo.z, blo.w, bhi.x, bhi.y, bhi.z, bhi.w};
    float v[8];
    unsigned int bits = 0u;
#pragma unroll
    for (int r = 0; r < 8; ++r) {
      v[r] = C[m][r] + (isH ? bb[r] : 0.0f);   // tangent gets no bias
      bits |= (v[r] > 0.0f ? 1u : 0u) << r;
    }
    unsigned int pbits = (unsigned int)__shfl_xor((int)bits, 8, 32);
    unsigned int use = isH ? bits : pbits;     // tangent uses primal's mask
#pragma unroll
    for (int r = 0; r < 8; ++r)
      C[m][r] = ((use >> r) & 1u) ? v[r] : SLOPE * v[r];
  }
}

// Convert activated f32 C tiles (M=128) into bf16 B k-tiles (K=128).
// C layout: lane half h holds sub-rows [8h,8h+8) of every m-tile.
// B layout: lane half h holds K = 32kb + 16h + (0..15) -> all from m-tile
// (2kb+h); the other sub-half's 8 values live in lane^16.
__device__ __forceinline__ void pack_exchange(const v8f C[8], int half,
                                              v16bf B[4]) {
  unsigned int pk[8][4];
#pragma unroll
  for (int m = 0; m < 8; ++m)
#pragma unroll
    for (int j = 0; j < 4; ++j)
      pk[m][j] = pack_bf16(C[m][2 * j], C[m][2 * j + 1]);
#pragma unroll
  for (int kb = 0; kb < 4; ++kb) {
    u32x8 u;
#pragma unroll
    for (int j = 0; j < 4; ++j) {
      unsigned int snd = half ? pk[2 * kb][j]     : pk[2 * kb + 1][j];
      unsigned int own = half ? pk[2 * kb + 1][j] : pk[2 * kb][j];
      unsigned int rcv = (unsigned int)__shfl_xor((int)snd, 16, 32);
      u[j]     = half ? rcv : own;   // dwords 0..3 = sub-rows 0..7
      u[j + 4] = half ? own : rcv;   // dwords 4..7 = sub-rows 8..15
    }
    B[kb] = __builtin_bit_cast(v16bf, u);
  }
}

// Prefetch one tile's global inputs into registers (primal lanes only).
__device__ __forceinline__ void prefetch_inputs(const float* __restrict__ x,
                                                int n, int d, int half,
                                                bool isH, float f[16],
                                                float* xxv) {
  const int b = n >> 8, w = n & 255;
  if (isH) {
    const float4* p = (const float4*)(x + (size_t)(b * 258 + w) * 16 + half * 16);
    float4 f0 = p[0], f1 = p[1], f2 = p[2], f3 = p[3];
    f[0] = f0.x; f[1] = f0.y; f[2]  = f0.z; f[3]  = f0.w;
    f[4] = f1.x; f[5] = f1.y; f[6]  = f1.z; f[7]  = f1.w;
    f[8] = f2.x; f[9] = f2.y; f[10] = f2.z; f[11] = f2.w;
    f[12] = f3.x; f[13] = f3.y; f[14] = f3.z; f[15] = f3.w;
    if (half == 0) *xxv = x[(size_t)(b * 258 + w + 2) * 16 + d];
  }
}

// ---------------------------------------------------------------------------
// grid = (64 groups of 128 samples, 16 latent dims); block = 128 = 4 waves.
// Each wave: 4 tiles of 8 samples (primal+tangent share the 16 B columns).
// ---------------------------------------------------------------------------
__global__ void __launch_bounds__(128)
np_prior_kernel(const float* __restrict__ x,
                const float* __restrict__ W0, const float* __restrict__ b0,
                const float* __restrict__ W1, const float* __restrict__ b1,
                const float* __restrict__ W2, const float* __restrict__ b2,
                const float* __restrict__ Wout, const float* __restrict__ bout,
                float* __restrict__ residuals, float* __restrict__ ws_logdet) {
  extern __shared__ __align__(16) char smem[];
  __bf16* sW0 = (__bf16*)smem;          // 128 x 72  (K padded 33->64, +8 bank pad)
  __bf16* sW1 = sW0 + 128 * 72;         // 128 x 136
  __bf16* sW2 = sW1 + 128 * 136;        // 128 x 136
  float*  sB0 = (float*)(sW2 + 128 * 136);
  float*  sB1 = sB0 + 128;
  float*  sB2 = sB1 + 128;
  float*  sWo = sB2 + 128;

  const int d = blockIdx.y;
  const int tid = threadIdx.x;

  // ---- stage weights fp32 -> bf16 into LDS ----
  {
    const float* W0s = W0 + (size_t)d * 128 * 33;
    for (int idx = tid; idx < 128 * 64; idx += 128) {
      int r = idx >> 6, c = idx & 63;
      sW0[r * 72 + c] = (c < 33) ? (__bf16)W0s[r * 33 + c] : (__bf16)0.0f;
    }
    const float* W1s = W1 + (size_t)d * 128 * 128;
    const float* W2s = W2 + (size_t)d * 128 * 128;
    for (int idx = tid; idx < 128 * 128; idx += 128) {
      int r = idx >> 7, c = idx & 127;
      sW1[r * 136 + c] = (__bf16)W1s[idx];
      sW2[r * 136 + c] = (__bf16)W2s[idx];
    }
    sB0[tid] = b0[(size_t)d * 128 + tid];
    sB1[tid] = b1[(size_t)d * 128 + tid];
    sB2[tid] = b2[(size_t)d * 128 + tid];
    sWo[tid] = Wout[(size_t)d * 128 + tid];
  }
  __syncthreads();

  const int lane = tid & 31;
  const int wave = tid >> 5;
  const int col  = lane & 15;        // B/C column
  const int half = lane >> 4;
  const bool isH = (col & 8) == 0;   // primal column vs tangent column
  const int scol = col & 7;          // sample index within tile
  const int nbase = blockIdx.x * 128 + wave * 32 + scol;
  const float bout_d = bout[d];

  const v8f zero = {0.f, 0.f, 0.f, 0.f, 0.f, 0.f, 0.f, 0.f};

  float fin[16];
  float xxv = 0.0f;
  prefetch_inputs(x, nbase, d, half, isH, fin, &xxv);

#pragma unroll 1
  for (int t = 0; t < 4; ++t) {
    const int n = nbase + t * 8;     // sample for data currently in fin/xxv

    // ---- build layer-0 B tiles from prefetched registers ----
    // primal col: K=0..31 = lag context, K=32 = x_d(t); tangent col: e_32
    v16bf B0t0, B0t1;
#pragma unroll
    for (int i = 0; i < 16; ++i) { B0t0[i] = (__bf16)0.0f; B0t1[i] = (__bf16)0.0f; }
    if (isH) {
#pragma unroll
      for (int i = 0; i < 16; ++i) B0t0[i] = (__bf16)fin[i];
    }
    if (half == 0)                                 // K=32 -> half 0, element 0
      B0t1[0] = isH ? (__bf16)xxv : (__bf16)1.0f;  // primal: x_d ; tangent: 1

    // kick off next tile's global loads; consumed only next iteration
    if (t < 3) prefetch_inputs(x, nbase + (t + 1) * 8, d, half, isH, fin, &xxv);

    // ---- layer 0: [128x64] x [64x16] ----
    v8f C[8];
#pragma unroll
    for (int m = 0; m < 8; ++m) {
      v16bf a0 = load_A(sW0, 72, m, 0,  col, half);
      v16bf a1 = load_A(sW0, 72, m, 32, col, half);
      C[m] = wmma_bf16(a1, B0t1, wmma_bf16(a0, B0t0, zero));
    }
    act_bias_mixed(C, sB0, half, isH);
    v16bf B[4];
    pack_exchange(C, half, B);

    // ---- layers 1 and 2: [128x128] x [128x16] ----
#pragma unroll 1
    for (int layer = 0; layer < 2; ++layer) {
      const __bf16* Wl = (layer == 0) ? sW1 : sW2;
#pragma unroll
      for (int m = 0; m < 8; ++m) {          // old C dead once B is packed:
        v8f c = zero;                        // write accumulators in place
#pragma unroll
        for (int kb = 0; kb < 4; ++kb)
          c = wmma_bf16(load_A(Wl, 136, m, kb * 32, col, half), B[kb], c);
        C[m] = c;
      }
      act_bias_mixed(C, (layer == 0) ? sB1 : sB2, half, isH);
      if (layer == 0) pack_exchange(C, half, B);
    }

    // ---- output: 1x128 dot (weights from LDS); cross-half reduce ----
    float acc = 0.f;
#pragma unroll
    for (int m = 0; m < 8; ++m) {
      const float* wp = sWo + m * 16 + half * 8;
      float4 w0v = *(const float4*)wp;
      float4 w1v = *(const float4*)(wp + 4);
      float wv[8] = {w0v.x, w0v.y, w0v.z, w0v.w, w1v.x, w1v.y, w1v.z, w1v.w};
#pragma unroll
      for (int r = 0; r < 8; ++r) acc += wv[r] * C[m][r];
    }
    acc += __shfl_xor(acc, 16, 32);
    if (half == 0) {
      if (isH) residuals[(size_t)n * 16 + d] = acc + bout_d;   // [B,256,D] flat
      else     ws_logdet[(size_t)n * 16 + d] = logf(fabsf(acc));
    }
  }
}

// Deterministic reduction over the 16 latent dims -> sum_log_abs_det_jacobian[N]
__global__ void __launch_bounds__(256)
sladj_reduce(const float* __restrict__ ws, float* __restrict__ out) {
  int n = blockIdx.x * blockDim.x + threadIdx.x;
  if (n < 8192) {
    const float* p = ws + (size_t)n * 16;
    float s = 0.f;
#pragma unroll
    for (int dd = 0; dd < 16; ++dd) s += p[dd];
    out[n] = s;
  }
}

extern "C" void kernel_launch(void* const* d_in, const int* in_sizes, int n_in,
                              void* d_out, int out_size, void* d_ws, size_t ws_size,
                              hipStream_t stream) {
  const float* x    = (const float*)d_in[0];
  const float* W0   = (const float*)d_in[1];
  const float* b0   = (const float*)d_in[2];
  const float* W1   = (const float*)d_in[3];
  const float* b1   = (const float*)d_in[4];
  const float* W2   = (const float*)d_in[5];
  const float* b2   = (const float*)d_in[6];
  const float* Wout = (const float*)d_in[7];
  const float* bout = (const float*)d_in[8];
  float* out = (float*)d_out;             // [131072 residuals][8192 sladj]
  float* ws  = (float*)d_ws;              // [8192 x 16] log|dout|

  const size_t shmem = (size_t)(128 * 72 + 2 * 128 * 136) * 2   // bf16 weights
                     + (size_t)(4 * 128) * 4;                   // biases + Wout
  (void)hipFuncSetAttribute((const void*)np_prior_kernel,
                            hipFuncAttributeMaxDynamicSharedMemorySize,
                            (int)shmem);

  dim3 grid(64, 16, 1), block(128, 1, 1);
  np_prior_kernel<<<grid, block, shmem, stream>>>(x, W0, b0, W1, b1, W2, b2,
                                                  Wout, bout, out, ws);
  sladj_reduce<<<32, 256, 0, stream>>>(ws, out + 32 * 256 * 16);
}
